// CausalSelfAttention_22024592293896
// MI455X (gfx1250) — matrix operand order repaired and checked by reading, not compiled, not run on hardware
//
#include <hip/hip_runtime.h>
#include <hip/hip_bf16.h>

// ---------------------------------------------------------------------------
// Problem constants (B=2, T=2048, C=1024, NH=16, HD=64)
// ---------------------------------------------------------------------------
#define BB   2
#define TT   2048
#define CC   1024
#define NHH  16
#define HDD  64
#define O3C  (3 * CC)          // 3072
#define MTOT (BB * TT)         // 4096 rows of x

typedef __attribute__((ext_vector_type(16))) __bf16 v16bf;
typedef __attribute__((ext_vector_type(8)))  float  v8f;

union Frag {            // one WMMA 16x32 bf16 operand (8 VGPRs / lane)
    v16bf v;
    uint4 u[2];
};

__device__ __forceinline__ unsigned short f2bf(float x) {
    unsigned int u = __builtin_bit_cast(unsigned int, x);
    u += 0x7FFFu + ((u >> 16) & 1u);          // round-to-nearest-even
    return (unsigned short)(u >> 16);
}
__device__ __forceinline__ float bf2f(unsigned short b) {
    unsigned int u = ((unsigned int)b) << 16;
    return __builtin_bit_cast(float, u);
}

// 32-bit LDS offset of a generic pointer that refers to __shared__ memory
__device__ __forceinline__ unsigned lds_off(const void* p) {
    return (unsigned)(unsigned long long)
        (__attribute__((address_space(3))) const unsigned char*)p;
}

// per-lane async copy: 16 bytes global -> LDS (ASYNCcnt-tracked)
__device__ __forceinline__ void async_cp16(void* dst_lds, const void* src) {
    asm volatile("global_load_async_to_lds_b128 %0, %1, off"
                 :: "v"(lds_off(dst_lds)), "v"((unsigned long long)src)
                 : "memory");
}
__device__ __forceinline__ void wait_async0() {
    asm volatile("s_wait_asynccnt 0x0" ::: "memory");
}

// ---------------------------------------------------------------------------
// K0: fp32 -> bf16 conversion of x (B*T*C) and w_attn (3C*C)
// ---------------------------------------------------------------------------
__global__ void cvt_bf16_kernel(const float* __restrict__ x,
                                const float* __restrict__ w,
                                unsigned short* __restrict__ xb,
                                unsigned short* __restrict__ wb) {
    const int NX = MTOT * CC;           // 4194304
    const int NW = O3C * CC;            // 3145728
    int i = blockIdx.x * 256 + threadIdx.x;
    if (i < NX) {
        xb[i] = f2bf(x[i]);
    } else {
        int j = i - NX;
        if (j < NW) wb[j] = f2bf(w[j]);
    }
}

// ---------------------------------------------------------------------------
// K1: qkv[m,o] = sum_k xb[m,k] * wb[o,k]   (M=4096, N=3072, K=1024)
// 128x128 output tile per 256-thread block (8 waves, 4x2).
// K chunked by 32; double-buffered 8KB A/B LDS tiles filled with
// global_load_async_to_lds_b128, fenced by s_wait_asynccnt + barrier.
// ---------------------------------------------------------------------------
__global__ __launch_bounds__(256) void qkv_gemm_kernel(
        const unsigned short* __restrict__ xb,
        const unsigned short* __restrict__ wb,
        unsigned short* __restrict__ qkv) {
    __shared__ unsigned short sA[2][128 * 32] __attribute__((aligned(16)));
    __shared__ unsigned short sB[2][128 * 32] __attribute__((aligned(16)));

    const int lane = threadIdx.x & 31;
    const int half = lane >> 4;                 // 0: lanes 0-15, 1: lanes 16-31
    const int ln   = lane & 15;
    const int wid  = threadIdx.x >> 5;          // 0..7
    const int wm   = wid >> 1;                  // 0..3 : 32-row strip
    const int wn   = wid & 1;                   // 0..1 : 64-col strip
    const int m0   = blockIdx.y * 128;
    const int n0   = blockIdx.x * 128;

    // stage one 128x32 A tile + 128x32 B tile into LDS buffer `buf`
    auto stage = [&](int kc, int buf) {
        const int seg = threadIdx.x & 3;        // which 16B of the 64B row
        const int row = threadIdx.x >> 2;       // 0..63
#pragma unroll
        for (int rs = 0; rs < 2; ++rs) {
            const int r = row + rs * 64;
            async_cp16(&sA[buf][r * 32 + seg * 8],
                       xb + (size_t)(m0 + r) * CC + kc + seg * 8);
            async_cp16(&sB[buf][r * 32 + seg * 8],
                       wb + (size_t)(n0 + r) * CC + kc + seg * 8);
        }
    };

    v8f acc[2][4] = {};

    stage(0, 0);
    wait_async0();
    __syncthreads();

    const int NCHUNK = CC / 32;                 // 32
#pragma unroll 1
    for (int c = 0; c < NCHUNK; ++c) {
        const int cur = c & 1;
        if (c + 1 < NCHUNK) stage((c + 1) * 32, cur ^ 1);

        // A fragments for this wave's two 16-row tiles (ISA 16-bit A layout)
        Frag aF[2];
#pragma unroll
        for (int mi = 0; mi < 2; ++mi) {
            const unsigned short* p =
                &sA[cur][(wm * 32 + mi * 16 + ln) * 32 + (half ? 8 : 0)];
            aF[mi].u[0] = *(const uint4*)(p);
            aF[mi].u[1] = *(const uint4*)(p + 16);
        }
#pragma unroll
        for (int j = 0; j < 4; ++j) {
            const unsigned short* p =
                &sB[cur][(wn * 64 + j * 16 + ln) * 32 + (half ? 8 : 0)];
            Frag bF;
            bF.u[0] = *(const uint4*)(p);
            bF.u[1] = *(const uint4*)(p + 16);
#pragma unroll
            for (int mi = 0; mi < 2; ++mi)
                acc[mi][j] = __builtin_amdgcn_wmma_f32_16x16x32_bf16(
                    false, aF[mi].v, false, bF.v, (short)0, acc[mi][j],
                    false, false);
        }

        wait_async0();      // this wave's prefetch into buf^1 landed
        __syncthreads();    // everyone done reading buf -> safe to overwrite
    }

    // C layout: VGPR r, lanes 0-15 -> M=r ; lanes 16-31 -> M=r+8 ; N = ln
#pragma unroll
    for (int mi = 0; mi < 2; ++mi) {
#pragma unroll
        for (int j = 0; j < 4; ++j) {
            const int col = n0 + wn * 64 + j * 16 + ln;
#pragma unroll
            for (int r = 0; r < 8; ++r) {
                const int row = m0 + wm * 32 + mi * 16 + r + (half ? 8 : 0);
                qkv[(size_t)row * O3C + col] = f2bf(acc[mi][j][r]);
            }
        }
    }
}

// ---------------------------------------------------------------------------
// K2: RoPE on q,k (in place, bf16) + V transpose to (b,h,d,t)
// ---------------------------------------------------------------------------
__global__ void rope_vt_kernel(unsigned short* __restrict__ qkv,
                               const float* __restrict__ fcos,
                               const float* __restrict__ fsin,
                               unsigned short* __restrict__ vt) {
    int tid = blockIdx.x * blockDim.x + threadIdx.x;     // 2^21 threads
    const int i = tid & 31;
    const int h = (tid >> 5) & 15;
    const int t = (tid >> 9) & (TT - 1);
    const int b = tid >> 20;
    if (b >= BB) return;

    const float cs = fcos[t * 32 + i];
    const float sn = fsin[t * 32 + i];
    const size_t base = (size_t)(b * TT + t) * O3C + h * HDD + 2 * i;

#pragma unroll
    for (int s = 0; s < 2; ++s) {                // s=0: q, s=1: k
        unsigned short* p = qkv + base + s * CC;
        const float tr = bf2f(p[0]);
        const float ti = bf2f(p[1]);
        p[0] = f2bf(tr * cs - ti * sn);
        p[1] = f2bf(tr * sn + ti * cs);
    }

    const unsigned short* vp = qkv + base + 2 * CC;
    const size_t vtb = ((size_t)((b * NHH + h) * HDD + 2 * i)) * TT + t;
    vt[vtb]      = vp[0];
    vt[vtb + TT] = vp[1];
}

// ---------------------------------------------------------------------------
// K3: flash attention. One wave per (b*h, 16-query tile).
// ---------------------------------------------------------------------------
__global__ __launch_bounds__(32) void flash_attn_kernel(
        const unsigned short* __restrict__ qkv,
        const unsigned short* __restrict__ vt,
        float* __restrict__ out) {
    __shared__ unsigned short sP[16 * 32] __attribute__((aligned(16)));

    const int lane = threadIdx.x & 31;
    const int half = lane >> 4;
    const int ln   = lane & 15;
    const int bh   = blockIdx.x;          // 0..31  (b*NH + h)
    const int qt   = blockIdx.y;          // 0..127 (16-row query tile)
    const int b = bh >> 4;
    const int h = bh & 15;

    Frag qf[2];
    {
        const int m = qt * 16 + ln;
        const unsigned short* qp = qkv + (size_t)(b * TT + m) * O3C + h * HDD;
#pragma unroll
        for (int f = 0; f < 2; ++f) {
            const unsigned short* p = qp + f * 32 + (half ? 8 : 0);
            qf[f].u[0] = *(const uint4*)(p);
            qf[f].u[1] = *(const uint4*)(p + 16);
        }
    }

    v8f oacc[4] = {};
    float mrun[8], lrun[8];
#pragma unroll
    for (int r = 0; r < 8; ++r) { mrun[r] = -1e30f; lrun[r] = 0.0f; }

    const int nkb = (qt + 2) >> 1;        // number of 32-key blocks (causal)

#pragma unroll 1
    for (int kb = 0; kb < nkb; ++kb) {
        const int k0 = kb * 32;

        v8f s[2];
#pragma unroll
        for (int hn = 0; hn < 2; ++hn) {
            const int key = k0 + hn * 16 + ln;
            const unsigned short* kp =
                qkv + (size_t)(b * TT + key) * O3C + CC + h * HDD;
            Frag k0f, k1f;
            const unsigned short* p0 = kp + (half ? 8 : 0);
            k0f.u[0] = *(const uint4*)(p0);
            k0f.u[1] = *(const uint4*)(p0 + 16);
            const unsigned short* p1 = kp + 32 + (half ? 8 : 0);
            k1f.u[0] = *(const uint4*)(p1);
            k1f.u[1] = *(const uint4*)(p1 + 16);
            v8f z = {};
            z = __builtin_amdgcn_wmma_f32_16x16x32_bf16(
                    false, qf[0].v, false, k0f.v, (short)0, z, false, false);
            z = __builtin_amdgcn_wmma_f32_16x16x32_bf16(
                    false, qf[1].v, false, k1f.v, (short)0, z, false, false);
            s[hn] = z;
        }

        float pl[2][8], alr[8];
#pragma unroll
        for (int r = 0; r < 8; ++r) {
            const int mrow = qt * 16 + r + (half ? 8 : 0);
#pragma unroll
            for (int hn = 0; hn < 2; ++hn) {
                const int keyn = k0 + hn * 16 + ln;
                float v = s[hn][r] * 0.125f;          // 1/sqrt(64)
                if (keyn > mrow) v = -1e30f;
                pl[hn][r] = v;
            }
        }
#pragma unroll
        for (int r = 0; r < 8; ++r) {
            float rm = fmaxf(pl[0][r], pl[1][r]);
            rm = fmaxf(rm, __shfl_xor(rm, 8, 32));
            rm = fmaxf(rm, __shfl_xor(rm, 4, 32));
            rm = fmaxf(rm, __shfl_xor(rm, 2, 32));
            rm = fmaxf(rm, __shfl_xor(rm, 1, 32));
            const float nm = fmaxf(mrun[r], rm);
            const float alpha = __expf(mrun[r] - nm);
            const float p0 = __expf(pl[0][r] - nm);
            const float p1 = __expf(pl[1][r] - nm);
            float rs = p0 + p1;
            rs += __shfl_xor(rs, 8, 32);
            rs += __shfl_xor(rs, 4, 32);
            rs += __shfl_xor(rs, 2, 32);
            rs += __shfl_xor(rs, 1, 32);
            lrun[r] = lrun[r] * alpha + rs;
            mrun[r] = nm;
            alr[r] = alpha;
            pl[0][r] = p0;
            pl[1][r] = p1;
        }
#pragma unroll
        for (int j = 0; j < 4; ++j)
#pragma unroll
            for (int r = 0; r < 8; ++r) oacc[j][r] *= alr[r];

        // P: C-layout -> A-layout via LDS (16x32 bf16 row-major)
#pragma unroll
        for (int hn = 0; hn < 2; ++hn)
#pragma unroll
            for (int r = 0; r < 8; ++r)
                sP[(r + (half ? 8 : 0)) * 32 + hn * 16 + ln] = f2bf(pl[hn][r]);
        __syncthreads();

        Frag pf;
        {
            const unsigned short* lp = sP + ln * 32 + (half ? 8 : 0);
            pf.u[0] = *(const uint4*)(lp);
            pf.u[1] = *(const uint4*)(lp + 16);
        }

#pragma unroll
        for (int j = 0; j < 4; ++j) {
            const int col = j * 16 + ln;
            const unsigned short* vp =
                vt + (size_t)((b * NHH + h) * HDD + col) * TT + k0 + (half ? 8 : 0);
            Frag vf;
            vf.u[0] = *(const uint4*)(vp);
            vf.u[1] = *(const uint4*)(vp + 16);
            oacc[j] = __builtin_amdgcn_wmma_f32_16x16x32_bf16(
                          false, pf.v, false, vf.v, (short)0, oacc[j], false, false);
        }
        __syncthreads();
    }

#pragma unroll
    for (int j = 0; j < 4; ++j) {
#pragma unroll
        for (int r = 0; r < 8; ++r) {
            const int row = qt * 16 + r + (half ? 8 : 0);
            out[((size_t)bh * TT + row) * HDD + j * 16 + ln] =
                oacc[j][r] / lrun[r];
        }
    }
}

// ---------------------------------------------------------------------------
// Launch
// ---------------------------------------------------------------------------
extern "C" void kernel_launch(void* const* d_in, const int* in_sizes, int n_in,
                              void* d_out, int out_size, void* d_ws, size_t ws_size,
                              hipStream_t stream) {
    (void)in_sizes; (void)n_in; (void)out_size;

    const float* x  = (const float*)d_in[0];
    const float* w  = (const float*)d_in[1];
    const float* fc = (const float*)d_in[2];
    const float* fs = (const float*)d_in[3];
    float* out = (float*)d_out;

    // Workspace: xb 8MB | wb 6MB(+2 pad) | qkv 24MB | vt 8MB  => 48MB
    if (ws_size < (48u << 20)) return;
    char* ws = (char*)d_ws;
    unsigned short* xb  = (unsigned short*)(ws);
    unsigned short* wb  = (unsigned short*)(ws + ((size_t)8  << 20));
    unsigned short* qkv = (unsigned short*)(ws + ((size_t)16 << 20));
    unsigned short* vt  = (unsigned short*)(ws + ((size_t)40 << 20));

    {
        const int total = MTOT * CC + O3C * CC;
        cvt_bf16_kernel<<<(total + 255) / 256, 256, 0, stream>>>(x, w, xb, wb);
    }
    // K1: 128x128 tiles -> grid (N/128, M/128) = (24, 32)
    qkv_gemm_kernel<<<dim3(O3C / 128, MTOT / 128), 256, 0, stream>>>(xb, wb, qkv);
    rope_vt_kernel<<<(1 << 21) / 256, 256, 0, stream>>>(qkv, fc, fs, vt);
    flash_attn_kernel<<<dim3(BB * NHH, TT / 16), 32, 0, stream>>>(qkv, vt, out);
}
